// CUTLASSSwiGLU2_54606214201650
// MI455X (gfx1250) — compile-verified, matrix operand order checked by reading, not run on hardware
//
#include <hip/hip_runtime.h>
#include <hip/hip_bf16.h>

// ---------------------------------------------------------------------------
// SwiGLU MLP with 2:4-sparse gate/up weight, targeted at MI455X (gfx1250).
//   GEMM1 (sparse, fused SiLU):  v_swmmac_f32_16x16x64_bf16, x tile staged in
//                                LDS by the Tensor Data Mover (tensor_load_to_lds,
//                                TENSORcnt), double-buffered.
//   GEMM2 (dense down-proj):     v_wmma_f32_16x16x32_bf16, x4 tile staged in
//                                LDS with global_load_async_to_lds_b128
//                                (ASYNCcnt), double-buffered.
// Wave32; each wave owns one 16x16 f32 accumulator tile (8 VGPRs) per chain.
// ---------------------------------------------------------------------------

#define TOKENS 8192   // 4 * 2048
#define KDIM   4096
#define MDIM   4096
#define KSTEP  128    // K elements staged per pipeline step (8 KB tile)

typedef __attribute__((ext_vector_type(16))) __bf16 v16bf;
typedef __attribute__((ext_vector_type(32))) __bf16 v32bf;
typedef __attribute__((ext_vector_type(8)))  float  v8f;
typedef __attribute__((ext_vector_type(4)))  float  f32x4;
typedef __attribute__((ext_vector_type(4)))  unsigned int u32x4;
typedef __attribute__((ext_vector_type(8)))  unsigned int u32x8;

union FragA { v16bf v; f32x4 q[2]; };   // 32 B: packed-sparse A 16x32 / dense 16x32 operand
union FragB { v32bf v; f32x4 q[4]; };   // 64 B: dense B 64x16 operand

static __device__ __forceinline__ unsigned short f32_to_bf16(float f) {
  unsigned int u = __float_as_uint(f);
  u += 0x7fffu + ((u >> 16) & 1u);       // round-to-nearest-even
  return (unsigned short)(u >> 16);
}

// Flat pointer -> LDS byte offset (LDS aperture keeps offset in addr[31:0]).
static __device__ __forceinline__ unsigned lds_off_of(const void* p) {
  return (unsigned)(unsigned long long)p;
}

// ---------------------------------------------------------------------------
// TDM: stage a [32 rows x 128 elems] bf16 tile (row stride = KDIM elems) from
// global into LDS. 2D tensor => descriptor groups 0 and 1 only.
// ---------------------------------------------------------------------------
static __device__ __forceinline__ void tdm_stage_tile(unsigned lds_off,
                                                      unsigned long long gaddr) {
  u32x4 g0;
  g0[0] = 1u;                                             // count=1, user D#
  g0[1] = lds_off;                                        // lds_addr (bytes)
  g0[2] = (unsigned)(gaddr & 0xffffffffull);              // global_addr[31:0]
  g0[3] = (unsigned)((gaddr >> 32) & 0x01ffffffull)       // global_addr[56:32]
          | 0x80000000u;                                  // type=2 ("image")
  u32x8 g1;
  g1[0] = 1u << 16;                                       // data_size=1 (2 B)
  g1[1] = ((unsigned)KDIM & 0xffffu) << 16;               // tensor_dim0[15:0]
  g1[2] = ((unsigned)KDIM >> 16)                          // tensor_dim0[31:16]
          | (((unsigned)TOKENS & 0xffffu) << 16);         // tensor_dim1[15:0]
  g1[3] = ((unsigned)TOKENS >> 16) | ((unsigned)KSTEP << 16); // dim1[31:16] | tile_dim0
  g1[4] = 32u;                                            // tile_dim1=32, tile_dim2=0
  g1[5] = (unsigned)KDIM;                                 // tensor_dim0_stride[31:0]
  g1[6] = 0u;
  g1[7] = 0u;
  asm volatile("tensor_load_to_lds %0, %1" :: "s"(g0), "s"(g1) : "memory");
}

// Per-lane async copy: 16 global bytes -> LDS (tracked by ASYNCcnt).
static __device__ __forceinline__ void async_b128_to_lds(const void* lds_dst,
                                                         const void* gsrc) {
  unsigned lo = lds_off_of(lds_dst);
  unsigned long long ga = (unsigned long long)gsrc;
  asm volatile("global_load_async_to_lds_b128 %0, %1, off"
               :: "v"(lo), "v"(ga) : "memory");
}
static __device__ __forceinline__ void wait_async_le2() {
  asm volatile("s_wait_asynccnt 0x2" ::: "memory");
}
static __device__ __forceinline__ void wait_async_0() {
  asm volatile("s_wait_asynccnt 0x0" ::: "memory");
}

// ---------------------------------------------------------------------------
// f32 -> bf16 bulk convert: 8 elements / thread (b128 in, b128 out)
// ---------------------------------------------------------------------------
__global__ void cvt_f32_to_bf16_kernel(const float* __restrict__ src,
                                       unsigned short* __restrict__ dst, int n8) {
  int i = blockIdx.x * blockDim.x + threadIdx.x;
  if (i >= n8) return;
  const f32x4* s = (const f32x4*)src + (size_t)i * 2;
  f32x4 a = s[0];
  f32x4 b = s[1];
  u32x4 o;
  o.x = (unsigned)f32_to_bf16(a.x) | ((unsigned)f32_to_bf16(a.y) << 16);
  o.y = (unsigned)f32_to_bf16(a.z) | ((unsigned)f32_to_bf16(a.w) << 16);
  o.z = (unsigned)f32_to_bf16(b.x) | ((unsigned)f32_to_bf16(b.y) << 16);
  o.w = (unsigned)f32_to_bf16(b.z) | ((unsigned)f32_to_bf16(b.w) << 16);
  *((u32x4*)dst + i) = o;
}

// ---------------------------------------------------------------------------
// Compress 2:4-sparse w12 [2M, K] f32 into packed bf16 [2M, K/2] + index words
// (u32 per 32 K: 8 groups x {idx0[1:0], idx1[3:2]}). Reference zeros are exact.
// ---------------------------------------------------------------------------
__global__ void pack_w12_kernel(const float* __restrict__ w12,
                                unsigned short* __restrict__ w12p,
                                unsigned int* __restrict__ w12idx) {
  int gid = blockIdx.x * blockDim.x + threadIdx.x;    // 2M * K/32 threads
  int row = gid >> 7;                                 // K/32 = 128 spans per row
  int gw  = gid & 127;
  const float* src = w12 + (size_t)row * KDIM + (size_t)gw * 32;
  unsigned short pk[16];
  unsigned int iw = 0;
#pragma unroll
  for (int g = 0; g < 8; ++g) {
    float vv[4];
    vv[0] = src[4 * g + 0]; vv[1] = src[4 * g + 1];
    vv[2] = src[4 * g + 2]; vv[3] = src[4 * g + 3];
    int i0 = -1, i1 = -1;
#pragma unroll
    for (int j = 0; j < 4; ++j)
      if (vv[j] != 0.0f) { if (i0 < 0) i0 = j; else if (i1 < 0) i1 = j; }
    if (i0 < 0)      { i0 = 0; i1 = 1; }
    else if (i1 < 0) { if (i0 == 3) { i0 = 2; i1 = 3; } else i1 = i0 + 1; }
    pk[2 * g + 0] = f32_to_bf16(vv[i0]);
    pk[2 * g + 1] = f32_to_bf16(vv[i1]);
    iw |= (unsigned)((i0 & 3) | ((i1 & 3) << 2)) << (4 * g);
  }
  u32x4* dst = (u32x4*)(w12p + (size_t)row * (KDIM / 2) + (size_t)gw * 16);
  u32x4 o0, o1;
  o0.x = pk[0] | ((unsigned)pk[1] << 16);   o0.y = pk[2]  | ((unsigned)pk[3]  << 16);
  o0.z = pk[4] | ((unsigned)pk[5] << 16);   o0.w = pk[6]  | ((unsigned)pk[7]  << 16);
  o1.x = pk[8] | ((unsigned)pk[9] << 16);   o1.y = pk[10] | ((unsigned)pk[11] << 16);
  o1.z = pk[12] | ((unsigned)pk[13] << 16); o1.w = pk[14] | ((unsigned)pk[15] << 16);
  dst[0] = o0; dst[1] = o1;
  w12idx[(size_t)row * (KDIM / 32) + gw] = iw;
}

// ---------------------------------------------------------------------------
// GEMM1 (2:4 sparse) + fused bias/SiLU/mul.
// D[m, n] = sum_k w12[m, k] * x[n, k];  m = feature, n = token.
// Block: 256 threads = 8 waves = 4 feature-tiles x 2 token-tiles over a
// 64-feature x 32-token region. x tile [32 tok x 128 K] staged in LDS by TDM,
// double-buffered; weights stream from global.
// ---------------------------------------------------------------------------
__global__ void __launch_bounds__(256)
sparse_gateup_silu_kernel(const unsigned short* __restrict__ w12p,
                          const unsigned int* __restrict__ w12idx,
                          const unsigned short* __restrict__ xb,
                          const float* __restrict__ b12,
                          unsigned short* __restrict__ x4) {
  __shared__ __align__(16) unsigned char sx[2][32 * KSTEP * 2];  // 2 x 8 KB
  const int lane  = threadIdx.x & 31;
  const int wave  = threadIdx.x >> 5;
  const int fblk  = blockIdx.x & 63;            // 64 feature blocks (64 feats)
  const int tblk  = blockIdx.x >> 6;            // 256 token blocks (32 tokens)
  const int m0    = fblk * 64 + (wave & 3) * 16;
  const int t0    = tblk * 32 + (wave >> 2) * 16;
  const int l16   = lane & 15;
  const int hi    = lane >> 4;

  const unsigned short* a1p = w12p + (size_t)(m0 + l16) * (KDIM / 2);
  const unsigned short* a2p = w12p + (size_t)(MDIM + m0 + l16) * (KDIM / 2);
  const unsigned int*   i1p = w12idx + (size_t)(m0 + l16) * (KDIM / 32);
  const unsigned int*   i2p = w12idx + (size_t)(MDIM + m0 + l16) * (KDIM / 32);
  const unsigned long long xrow =
      (unsigned long long)(const void*)(xb + (size_t)(tblk * 32) * KDIM);

  if (wave == 0) tdm_stage_tile(lds_off_of(&sx[0][0]), xrow);

  v8f acc1 = {};   // gate
  v8f acc2 = {};   // up
  const int NSTEP = KDIM / KSTEP;               // 32
  for (int i = 0; i < NSTEP; ++i) {
    const int kk = i * KSTEP;
    if (wave == 0) {
      if (i + 1 < NSTEP) {
        tdm_stage_tile(lds_off_of(&sx[(i + 1) & 1][0]), xrow + 2ull * (kk + KSTEP));
        __builtin_amdgcn_s_wait_tensorcnt(1);   // current buffer staged
      } else {
        __builtin_amdgcn_s_wait_tensorcnt(0);
      }
    }
    __syncthreads();                            // tile visible to all waves
    const unsigned char* brow =
        sx[i & 1] + ((wave >> 2) * 16 + l16) * (KSTEP * 2);
#pragma unroll
    for (int sub = 0; sub < KSTEP; sub += 64) {
      const int kc = (kk + sub) >> 1;           // compressed K base
      FragA a1, a2;
      FragB b;
      a1.q[0] = *(const f32x4*)(a1p + kc + hi * 8);
      a1.q[1] = *(const f32x4*)(a1p + kc + 16 + hi * 8);
      a2.q[0] = *(const f32x4*)(a2p + kc + hi * 8);
      a2.q[1] = *(const f32x4*)(a2p + kc + 16 + hi * 8);
      int idx1 = (int)i1p[((kk + sub) >> 5) + hi];
      int idx2 = (int)i2p[((kk + sub) >> 5) + hi];
      const unsigned char* bb = brow + sub * 2;           // ds_load_b128 x4
      b.q[0] = *(const f32x4*)(bb + hi * 32);
      b.q[1] = *(const f32x4*)(bb + hi * 32 + 16);
      b.q[2] = *(const f32x4*)(bb + 64 + hi * 32);
      b.q[3] = *(const f32x4*)(bb + 64 + hi * 32 + 16);
      __builtin_prefetch(a1p + kc + 128, 0, 0);           // global_prefetch_b8
      __builtin_prefetch(a2p + kc + 128, 0, 0);
      acc1 = __builtin_amdgcn_swmmac_f32_16x16x64_bf16(
          false, a1.v, false, b.v, acc1, idx1, false, false);
      acc2 = __builtin_amdgcn_swmmac_f32_16x16x64_bf16(
          false, a2.v, false, b.v, acc2, idx2, false, false);
    }
    __syncthreads();                            // reads done before overwrite
  }

  // Epilogue: C/D layout — VGPR r, lane l: M = r + hi*8, N = l16.
#pragma unroll
  for (int r = 0; r < 8; ++r) {
    const int feat = m0 + r + hi * 8;
    const int tok  = t0 + l16;
    float g = acc1[r] + b12[feat];
    float u = acc2[r] + b12[MDIM + feat];
    float s = g / (1.0f + __expf(-g));          // SiLU
    x4[(size_t)tok * MDIM + feat] = f32_to_bf16(s * u);
  }
}

// ---------------------------------------------------------------------------
// GEMM2 (dense): out[n_out, tok] = sum_m w3[n_out, m] * x4[tok, m] + b3.
// Same block shape; x4 tile staged with per-lane async loads to LDS
// (each thread copies 2 x 16 B of its token row), double-buffered.
// ---------------------------------------------------------------------------
__global__ void __launch_bounds__(256)
dense_down_kernel(const unsigned short* __restrict__ w3b,
                  const unsigned short* __restrict__ x4,
                  const float* __restrict__ b3,
                  float* __restrict__ out) {
  __shared__ __align__(16) unsigned char sx[2][32 * KSTEP * 2];  // 2 x 8 KB
  const int lane  = threadIdx.x & 31;
  const int wave  = threadIdx.x >> 5;
  const int nblk  = blockIdx.x & 63;
  const int tblk  = blockIdx.x >> 6;
  const int n0    = nblk * 64 + (wave & 3) * 16;
  const int t0    = tblk * 32 + (wave >> 2) * 16;
  const int l16   = lane & 15;
  const int hi    = lane >> 4;

  const unsigned short* ap = w3b + (size_t)(n0 + l16) * MDIM;
  // Staging decomposition: thread -> (token = tid/8, 16B-chunk = tid%8).
  const int stok = threadIdx.x >> 3;
  const int schk = threadIdx.x & 7;
  const unsigned short* gx =
      x4 + (size_t)(tblk * 32 + stok) * MDIM + (size_t)schk * 8;
  unsigned char* const srow0 = &sx[0][stok * (KSTEP * 2) + schk * 16];
  unsigned char* const srow1 = &sx[1][stok * (KSTEP * 2) + schk * 16];

  // First stage (buffer 0, kk = 0): 2 async b128 per lane => ASYNCcnt = 2.
  async_b128_to_lds(srow0, gx);
  async_b128_to_lds(srow0 + 128, gx + 64);

  v8f acc = {};
  const int NSTEP = MDIM / KSTEP;               // 32
  for (int i = 0; i < NSTEP; ++i) {
    const int kk = i * KSTEP;
    if (i + 1 < NSTEP) {
      unsigned char* sn = ((i + 1) & 1) ? srow1 : srow0;
      async_b128_to_lds(sn, gx + kk + KSTEP);
      async_b128_to_lds(sn + 128, gx + kk + KSTEP + 64);
      wait_async_le2();                         // current buffer's 2 done
    } else {
      wait_async_0();
    }
    __syncthreads();
    const unsigned char* brow =
        sx[i & 1] + ((wave >> 2) * 16 + l16) * (KSTEP * 2);
#pragma unroll
    for (int sub = 0; sub < KSTEP; sub += 32) {
      FragA a, b;
      a.q[0] = *(const f32x4*)(ap + kk + sub + hi * 8);
      a.q[1] = *(const f32x4*)(ap + kk + sub + 16 + hi * 8);
      const unsigned char* bb = brow + sub * 2;           // ds_load_b128 x2
      b.q[0] = *(const f32x4*)(bb + hi * 16);
      b.q[1] = *(const f32x4*)(bb + 32 + hi * 16);
      __builtin_prefetch(ap + kk + sub + 128, 0, 0);
      acc = __builtin_amdgcn_wmma_f32_16x16x32_bf16(
          false, a.v, false, b.v, (short)0, acc, false, false);
    }
    __syncthreads();
  }

#pragma unroll
  for (int r = 0; r < 8; ++r) {
    const int nf  = n0 + r + hi * 8;
    const int tok = t0 + l16;
    out[(size_t)tok * MDIM + nf] = acc[r] + b3[nf];
  }
}

// ---------------------------------------------------------------------------
extern "C" void kernel_launch(void* const* d_in, const int* in_sizes, int n_in,
                              void* d_out, int out_size, void* d_ws, size_t ws_size,
                              hipStream_t stream) {
  (void)in_sizes; (void)n_in; (void)out_size; (void)ws_size;
  const float* x   = (const float*)d_in[0];   // [8192, 4096]
  const float* w12 = (const float*)d_in[1];   // [8192, 4096] 2:4 sparse
  const float* b12 = (const float*)d_in[2];   // [8192]
  const float* w3  = (const float*)d_in[3];   // [4096, 4096]
  const float* b3  = (const float*)d_in[4];   // [4096]
  float* out = (float*)d_out;                 // [8192, 4096]

  char* ws = (char*)d_ws;
  unsigned short* xb   = (unsigned short*)ws; ws += (size_t)TOKENS * KDIM * 2;         // 64 MiB
  unsigned short* w3b  = (unsigned short*)ws; ws += (size_t)MDIM * MDIM * 2;           // 32 MiB
  unsigned short* w12p = (unsigned short*)ws; ws += (size_t)2 * MDIM * (KDIM / 2) * 2; // 32 MiB
  unsigned int*   w12i = (unsigned int*)ws;   ws += (size_t)2 * MDIM * (KDIM / 32) * 4;// 4 MiB
  unsigned short* x4   = (unsigned short*)ws;                                          // 64 MiB

  // Pre-passes: bf16 conversion + 2:4 compression.
  cvt_f32_to_bf16_kernel<<<(TOKENS * (KDIM / 8)) / 256, 256, 0, stream>>>(x, xb, TOKENS * KDIM / 8);
  cvt_f32_to_bf16_kernel<<<(MDIM * (MDIM / 8)) / 256, 256, 0, stream>>>(w3, w3b, MDIM * MDIM / 8);
  pack_w12_kernel<<<(2 * MDIM * (KDIM / 32)) / 256, 256, 0, stream>>>(w12, w12p, w12i);

  // GEMM1: 64 feature-blocks x 256 token-blocks.
  sparse_gateup_silu_kernel<<<64 * 256, 256, 0, stream>>>(w12p, w12i, xb, b12, x4);

  // GEMM2: 64 out-blocks x 256 token-blocks.
  dense_down_kernel<<<64 * 256, 256, 0, stream>>>(w3b, x4, b3, out);
}
